// GraphConvolution_53145925320744
// MI455X (gfx1250) — matrix operand order repaired and checked by reading, not compiled, not run on hardware
//
#include <hip/hip_runtime.h>

// MI455X / gfx1250, wave32. Dominant cost: (T*d) @ T^T, 4096x4096x8192 =
// 275 GFLOP -> bf16 WMMA (16x16x32). Fused with adj_v/diag and the second
// GEMM (@HW) so the 4096x4096 multiplier is never materialized.
// Phase-1 tiles are DMA'd by the Tensor Data Mover (tensor_load_to_lds),
// software-pipelined against the WMMAs via TENSORcnt; f32->bf16 uses the
// native packed converts.

#define DEVINL static __device__ __forceinline__

typedef __attribute__((ext_vector_type(16))) __bf16 v16bf;
typedef __attribute__((ext_vector_type(8)))  float  v8f;
typedef __attribute__((ext_vector_type(4)))  unsigned u32x4;
typedef __attribute__((ext_vector_type(8)))  unsigned u32x8;

constexpr int NV = 4096, NE = 8192, IN_V = 128, OUT_V = 128, IN_E = 64;

DEVINL __bf16 f2bf(float f) { return (__bf16)f; }   // -> v_cvt_pk_bf16_f32

DEVINL v8f zero8() {
  v8f z;
  #pragma unroll
  for (int i = 0; i < 8; ++i) z[i] = 0.0f;
  return z;
}

DEVINL v8f wmma_bf16(v16bf a, v16bf b, v8f c) {
  return __builtin_amdgcn_wmma_f32_16x16x32_bf16(false, a, false, b,
                                                 (short)0, c, false, false);
}

// ---- Tensor Data Mover: 2D tile load, D# per CDNA5 ISA.
// group0: count=1 | lds_addr | global_addr[56:0] | type=2
// group1: data_size=4B, tensor_dim0/1, tile_dim0/1, tensor_dim0_stride
DEVINL void tdm_load_2d(unsigned lds_off, const void* gptr,
                        unsigned tile_d0, unsigned tile_d1,
                        unsigned tensor_d0, unsigned tensor_d1,
                        unsigned stride0) {
  unsigned long long ga = (unsigned long long)(size_t)gptr;
  u32x4 g0;
  g0[0] = 1u;                                   // count=1 (valid user D#)
  g0[1] = lds_off;                              // lds_addr (bytes)
  g0[2] = (unsigned)ga;                         // global_addr[31:0]
  g0[3] = (unsigned)(ga >> 32) | 0x80000000u;   // global_addr[56:32] | type=2
  u32x8 g1;
  g1[0] = 0x00020000u;                          // workgroup_mask=0, data_size=4B
  g1[1] = (tensor_d0 & 0xFFFFu) << 16;          // tensor_dim0[15:0] @ [63:48]
  g1[2] = (tensor_d0 >> 16) | ((tensor_d1 & 0xFFFFu) << 16);
  g1[3] = (tensor_d1 >> 16) | (tile_d0 << 16);  // tile_dim0 @ [127:112]
  g1[4] = tile_d1 & 0xFFFFu;                    // tile_dim1 (tile_dim2=0)
  g1[5] = stride0;                              // tensor_dim0_stride[31:0]
  g1[6] = 0u;
  g1[7] = 0u;
  asm volatile("tensor_load_to_lds %0, %1" :: "s"(g0), "s"(g1) : "memory");
}

// A-operand (16x32 MxK) from row-major [m][k] bf16 in LDS.
// Lane l: m=l&15 ; K=(l>>4)*16 + 0..15 contiguous.
DEVINL v16bf afrag_lds(const __bf16* base, int ld) {
  int l = threadIdx.x & 31;
  const __bf16* p = base + (l & 15) * ld + ((l >> 4) << 4);
  union { uint4 u[2]; v16bf v; } c;
  c.u[0] = *(const uint4*)(p);
  c.u[1] = *(const uint4*)(p + 8);
  return c.v;
}

// A-operand from row-major f32 (global), converted to bf16.
DEVINL v16bf afrag_f32(const float* base, int ld) {
  int l = threadIdx.x & 31;
  const float* p = base + (l & 15) * ld + ((l >> 4) << 4);
  v16bf r;
  #pragma unroll
  for (int i = 0; i < 8; ++i) r[i] = f2bf(p[i]);
  #pragma unroll
  for (int i = 0; i < 8; ++i) r[8 + i] = f2bf(p[8 + i]);
  return r;
}

// B-operand (32x16 KxN) where the source holds B^T row-major [n][k] (bf16,
// LDS or global). Lane l: n=l&15 ; K=(l>>4)*16 + 0..15 contiguous.
DEVINL v16bf bfragT(const __bf16* base, int ld) {
  int l = threadIdx.x & 31;
  const __bf16* p = base + (size_t)(l & 15) * ld + ((l >> 4) << 4);
  union { uint4 u[2]; v16bf v; } c;
  c.u[0] = *(const uint4*)(p);
  c.u[1] = *(const uint4*)(p + 8);
  return c.v;
}

// B-operand from row-major [k][n] f32 in global (strided gather, small use).
DEVINL v16bf bfrag_gather_f32(const float* base, int ld, int col) {
  int l = threadIdx.x & 31;
  int c = col + (l & 15);
  int kb = (l >> 4) << 4;
  v16bf r;
  #pragma unroll
  for (int i = 0; i < 16; ++i) r[i] = f2bf(base[(size_t)(kb + i) * ld + c]);
  return r;
}

DEVINL void st4bf(__bf16* p, float4 a) {
  union { __bf16 h[4]; ushort4 u; } c;
  c.h[0] = f2bf(a.x); c.h[1] = f2bf(a.y); c.h[2] = f2bf(a.z); c.h[3] = f2bf(a.w);
  *(ushort4*)p = c.u;
}

// ---------------------------------------------------------------- d = H_e @ p
__global__ void k_dvec(const float* __restrict__ He, const float* __restrict__ p,
                       float* __restrict__ dv) {
  int i = blockIdx.x * blockDim.x + threadIdx.x;
  const float4* row = (const float4*)(He + (size_t)i * IN_E);
  const float4* pv  = (const float4*)p;
  float acc = 0.0f;
  #pragma unroll
  for (int q = 0; q < IN_E / 4; ++q) {
    float4 a = row[q], b = pv[q];
    acc += a.x * b.x + a.y * b.y + a.z * b.z + a.w * b.w;
  }
  dv[i] = acc;
}

// ------------------------------------------------------- H_e passthrough copy
__global__ void k_copy(const float4* __restrict__ src, float4* __restrict__ dst) {
  int i = blockIdx.x * blockDim.x + threadIdx.x;
  dst[i] = src[i];
}

// -------------------- HWT = bf16((H_v @ weight)^T), stored [OUT_V][NV] so the
// fused kernel's B-operands are contiguous 16B loads.
__global__ void k_hw(const float* __restrict__ Hv, const float* __restrict__ W,
                     __bf16* __restrict__ HWT) {
  int wave = threadIdx.x >> 5;
  int g = blockIdx.x * (blockDim.x >> 5) + wave;   // 0..2047
  int i0 = (g >> 3) * 16;
  int c0 = (g & 7) * 16;
  int l = threadIdx.x & 31;
  v8f acc = zero8();
  #pragma unroll
  for (int kc = 0; kc < IN_V / 32; ++kc) {
    v16bf a = afrag_f32(Hv + (size_t)i0 * IN_V + kc * 32, IN_V);
    v16bf b = bfrag_gather_f32(W + (size_t)kc * 32 * OUT_V, OUT_V, c0);
    acc = wmma_bf16(a, b, acc);
  }
  int n = l & 15, mb = (l >> 4) << 3;
  #pragma unroll
  for (int v = 0; v < 8; ++v)
    HWT[(size_t)(c0 + n) * NV + i0 + mb + v] = f2bf(acc[v]);
}

// ----------------------------------------------------------- fused main kernel
// out[i,:] = sum_j ((i==j ? 1 : S[i,j]) * adj_v[i,j]) * HW[j,:] + bias
// with S = (T*d) @ T^T.   Grid: NV/128 workgroups x 256 threads (8 waves).
//
// LDS map (64 KB):
//   [    0,16384) stgA   f32 TDM staging, (T*d) source tile 128x32
//   [16384,32768) stgB   f32 TDM staging, T source tile 128x32
//   [32768,40960) opA0 \
//   [40960,49152) opB0  \ double-buffered bf16 operand tiles (phase 1)
//   [49152,57344) opA1  /
//   [57344,65536) opB1 /
//   [32768,65536) lS    (S.*adj) bf16 block (phase 2; aliases op*, dead then)
__global__ void __launch_bounds__(256) k_main(
    const float* __restrict__ T, const float* __restrict__ dv,
    const float* __restrict__ adj, const __bf16* __restrict__ HWT,
    const float* __restrict__ bias, float* __restrict__ out) {
  __shared__ __align__(16) char smem[65536];
  float* stgA = (float*)smem;
  float* stgB = (float*)(smem + 16384);
  __bf16* lS  = (__bf16*)(smem + 32768);

  const int i0 = blockIdx.x * 128;
  const int w  = threadIdx.x >> 5;                 // wave -> i rows 16w..16w+15
  const int l  = threadIdx.x & 31;
  const int srow  = threadIdx.x >> 1;              // staging row 0..127
  const int shalf = (threadIdx.x & 1) * 16;        // staging col base 0|16

  const unsigned offA = (unsigned)(size_t)(void*)stgA;
  const unsigned offB = (unsigned)(size_t)(void*)stgB;

  v8f Oacc[8];
  #pragma unroll
  for (int t = 0; t < 8; ++t) Oacc[t] = zero8();

  for (int jb = 0; jb < NV / 128; ++jb) {
    const int j0 = jb * 128;
    v8f S[8];
    #pragma unroll
    for (int t = 0; t < 8; ++t) S[t] = zero8();

    // ---- Phase 1 prologue: DMA chunk 0 into staging
    if (w == 0) {
      tdm_load_2d(offA, T + (size_t)i0 * NE, 32, 128, NE, NV, NE);
      tdm_load_2d(offB, T + (size_t)j0 * NE, 32, 128, NE, NV, NE);
      __builtin_amdgcn_s_wait_tensorcnt(0);
    }
    __syncthreads();

    // ---- Phase 1: S(128x128) += (T*d)[i-block] @ T[j-block]^T over K=8192
    for (int kc = 0; kc < NE / 32; ++kc) {
      const int k0 = kc * 32;
      // ping-pong operand buffers (computed arithmetically, no LDS ptr arrays)
      __bf16* la = (__bf16*)(smem + 32768 + ((kc & 1) << 14));
      __bf16* lb = (__bf16*)(smem + 40960 + ((kc & 1) << 14));
      {  // convert staging -> bf16 operand tiles (A scaled by d)
        const float4* sAf = (const float4*)(stgA + srow * 32 + shalf);
        const float4* sBf = (const float4*)(stgB + srow * 32 + shalf);
        const float4* gd  = (const float4*)(dv + k0 + shalf);
        __bf16* sa = la + srow * 32 + shalf;
        __bf16* sb = lb + srow * 32 + shalf;
        #pragma unroll
        for (int q = 0; q < 4; ++q) {
          float4 a = sAf[q], b = sBf[q], d = gd[q];
          a.x *= d.x; a.y *= d.y; a.z *= d.z; a.w *= d.w;
          st4bf(sa + q * 4, a);
          st4bf(sb + q * 4, b);
        }
      }
      __syncthreads();                 // staging consumed; operand tiles ready

      const bool more = (kc + 1 < NE / 32);
      if (w == 0 && more) {            // async DMA of next chunk under the WMMAs
        tdm_load_2d(offA, T + (size_t)i0 * NE + k0 + 32, 32, 128, NE, NV, NE);
        tdm_load_2d(offB, T + (size_t)j0 * NE + k0 + 32, 32, 128, NE, NV, NE);
      }
      {
        v16bf af = afrag_lds(la + w * 16 * 32, 32);
        #pragma unroll
        for (int jt = 0; jt < 8; ++jt) {
          v16bf bf_ = bfragT(lb + jt * 16 * 32, 32);
          S[jt] = wmma_bf16(af, bf_, S[jt]);
        }
      }
      if (w == 0 && more) __builtin_amdgcn_s_wait_tensorcnt(0);
      __syncthreads();                 // next chunk staged; op[(kc+1)&1] free
    }

    // ---- Phase 2a: combine with adj_v (+diag fix), re-layout via LDS as bf16
    {
      const int n = l & 15, mb = (l >> 4) << 3;
      if (jb + 1 < NV / 128)                       // warm L2 for next adj strip
        __builtin_prefetch(adj + (size_t)(i0 + w * 16 + mb) * NV + j0 + 128 + n,
                           0, 1);
      #pragma unroll
      for (int jt = 0; jt < 8; ++jt) {
        int gj = j0 + jt * 16 + n;
        #pragma unroll
        for (int v = 0; v < 8; ++v) {
          int ri = w * 16 + mb + v;                // row within i-block
          int gi = i0 + ri;
          float m1 = (gi == gj) ? 1.0f : S[jt][v];
          float val = m1 * adj[(size_t)gi * NV + gj];
          lS[ri * 128 + jt * 16 + n] = f2bf(val);
        }
      }
    }
    __syncthreads();

    // ---- Phase 2b: Oacc(16x128 per wave) += (S.*adj) @ HW[j-block]
    #pragma unroll
    for (int kc2 = 0; kc2 < 4; ++kc2) {
      v16bf af = afrag_lds(lS + w * 16 * 128 + kc2 * 32, 128);
      #pragma unroll
      for (int ct = 0; ct < 8; ++ct) {
        v16bf bf_ = bfragT(HWT + (size_t)(ct * 16) * NV + j0 + kc2 * 32, NV);
        Oacc[ct] = wmma_bf16(af, bf_, Oacc[ct]);
      }
    }
    __syncthreads();   // lS reads done before next j-block reuses the region
  }

  // ---- Epilogue: + bias, store f32
  {
    const int n = l & 15, mb = (l >> 4) << 3;
    #pragma unroll
    for (int ct = 0; ct < 8; ++ct) {
      int c = ct * 16 + n;
      float bv = bias[c];
      #pragma unroll
      for (int v = 0; v < 8; ++v) {
        int gi = i0 + w * 16 + mb + v;
        out[(size_t)gi * OUT_V + c] = Oacc[ct][v] + bv;
      }
    }
  }
}

// ------------------------------------------------------------------ launcher
extern "C" void kernel_launch(void* const* d_in, const int* in_sizes, int n_in,
                              void* d_out, int out_size, void* d_ws, size_t ws_size,
                              hipStream_t stream) {
  (void)in_sizes; (void)n_in; (void)out_size; (void)ws_size;
  const float* H_v    = (const float*)d_in[0];
  const float* H_e    = (const float*)d_in[1];
  // d_in[2] = adj_e (unused by node_layer path)
  const float* adj_v  = (const float*)d_in[3];
  const float* T      = (const float*)d_in[4];
  const float* weight = (const float*)d_in[5];
  const float* bias   = (const float*)d_in[6];
  const float* p      = (const float*)d_in[7];

  float* out = (float*)d_out;                       // [NV*OUT_V] then [NE*IN_E]

  float*  dv  = (float*)d_ws;                       // 8192 f32 = 32 KB
  __bf16* HWT = (__bf16*)((char*)d_ws + 32768);     // [OUT_V][NV] bf16 = 1 MB

  k_dvec<<<NE / 256, 256, 0, stream>>>(H_e, p, dv);
  k_copy<<<(NE * IN_E / 4) / 256, 256, 0, stream>>>(
      (const float4*)H_e, (float4*)(out + (size_t)NV * OUT_V));
  k_hw<<<(NV / 16) * (OUT_V / 16) / 8, 256, 0, stream>>>(H_v, weight, HWT);
  k_main<<<NV / 128, 256, 0, stream>>>(T, dv, adj_v, HWT, bias, out);
}
